// EdgeDecoder_16638703305284
// MI455X (gfx1250) — matrix operand order repaired and checked by reading, not compiled, not run on hardware
//
#include <hip/hip_runtime.h>
#include <hip/hip_bf16.h>
#include <math.h>

#define HID 128

typedef __bf16 v16bf __attribute__((ext_vector_type(16)));
typedef float  v8f   __attribute__((ext_vector_type(8)));

struct BFrag { unsigned int u[8]; };   // 16 bf16 = 8 dwords = one WMMA operand per lane

__device__ __forceinline__ float bf2f(unsigned int bits16) {
  union { unsigned int u; float f; } c; c.u = bits16 << 16; return c.f;
}
__device__ __forceinline__ unsigned short f2bf(float f) {
  union { float f; unsigned int u; } c; c.f = f;
  unsigned int u = c.u;
  unsigned int r = u + 0x7FFFu + ((u >> 16) & 1u);   // round-to-nearest-even
  return (unsigned short)(r >> 16);
}
// packed bf16x2 multiply on the CDNA5 VOP3P pipe (RNE, denorm-preserving)
__device__ __forceinline__ unsigned int pk_mul_bf(unsigned int a, unsigned int b) {
  unsigned int d;
  asm("v_pk_mul_bf16 %0, %1, %2" : "=v"(d) : "v"(a), "v"(b));
  return d;
}

// ---------------------------------------------------------------- phase 0
__global__ void f32_to_bf16_kernel(const float* __restrict__ src,
                                   unsigned short* __restrict__ dst, int n) {
  int i = blockIdx.x * blockDim.x + threadIdx.x;
  int stride = gridDim.x * blockDim.x;
  for (; i < n; i += stride) dst[i] = f2bf(src[i]);
}

// ---------------------------------------------------------------- phase 1
// Fused per-node GEMMs sharing one A-fragment load of z:
//   t  [n][d] = sum_k z[n][k] * w_b [d*128+k]
//   a1 [n][h] = sum_k z[n][k] * w_c1[h*256+k]
//   a2 [n][h] = sum_k z[n][k] * w_c1[h*256+128+k]
// One wave = 16-node tile; N=128 in 8 n-tiles; K=128 in 4 WMMA steps; x3 matrices.
__global__ void node_gemm3_bf16_kernel(const unsigned short* __restrict__ zbf,
                                       const unsigned short* __restrict__ wbbf,
                                       const unsigned short* __restrict__ wc1bf,
                                       unsigned short* __restrict__ tbf,
                                       unsigned short* __restrict__ a1bf,
                                       unsigned short* __restrict__ a2bf,
                                       int ntiles, int nodes) {
  int wave = blockIdx.x * (blockDim.x >> 5) + (threadIdx.x >> 5);
  wave = __builtin_amdgcn_readfirstlane(wave);     // provably wave-uniform
  if (wave >= ntiles) return;
  int lane = threadIdx.x & 31;
  int r = lane & 15, half = lane >> 4;
  long m0 = (long)wave * 16;
  bool full = (m0 + 16 <= (long)nodes);            // scalar full-tile test
  long node = m0 + r; if (node >= nodes) node = nodes - 1;

  // A fragment: 16x32 bf16 tile of z rows (ISA 16-bit A lane layout)
  const unsigned int* zrow = (const unsigned int*)(zbf + node * HID);
  BFrag A[4];
  #pragma unroll
  for (int kt = 0; kt < 4; ++kt)
    #pragma unroll
    for (int v = 0; v < 8; ++v) {
      int k = kt * 32 + ((v < 4) ? (2 * v) : (16 + 2 * (v - 4))) + half * 8;
      A[kt].u[v] = zrow[k >> 1];
    }

  const unsigned short* Ws[3]  = {wbbf, wc1bf, wc1bf};
  const int             ldw_[3] = {HID, 2 * HID, 2 * HID};
  const int             koff_[3] = {0, 0, HID};
  unsigned short*       Os[3]  = {tbf, a1bf, a2bf};

  #pragma unroll
  for (int mtx = 0; mtx < 3; ++mtx) {
    const unsigned short* wbf = Ws[mtx];
    unsigned short* outbf = Os[mtx];
    int ldw = ldw_[mtx], koff = koff_[mtx];
    #pragma unroll
    for (int nt = 0; nt < 8; ++nt) {
      int n = nt * 16 + r;                         // output column / weight row
      const unsigned int* wrow = (const unsigned int*)(wbf + (long)n * ldw + koff);
      BFrag Bf[4];
      #pragma unroll
      for (int kt = 0; kt < 4; ++kt)
        #pragma unroll
        for (int v = 0; v < 8; ++v) {              // B: lanes0-15 K=0..7,16..23, lanes16-31 K=8..15,24..31
          int k = kt * 32 + ((v < 4) ? (2 * v) : (16 + 2 * (v - 4))) + half * 8;
          Bf[kt].u[v] = wrow[k >> 1];
        }
      v8f acc = {0.f, 0.f, 0.f, 0.f, 0.f, 0.f, 0.f, 0.f};
      #pragma unroll
      for (int kt = 0; kt < 4; ++kt) {
        v16bf av = __builtin_bit_cast(v16bf, A[kt]);
        v16bf bv = __builtin_bit_cast(v16bf, Bf[kt]);
        acc = __builtin_amdgcn_wmma_f32_16x16x32_bf16(false, av, false, bv,
                                                      (short)0, acc, false, false);
      }
      if (full) {
        #pragma unroll
        for (int rr = 0; rr < 8; ++rr)
          outbf[(m0 + rr + half * 8) * HID + n] = f2bf(acc[rr]);
      } else {
        #pragma unroll
        for (int rr = 0; rr < 8; ++rr) {
          long m = m0 + rr + half * 8;
          if (m < nodes) outbf[m * HID + n] = f2bf(acc[rr]);
        }
      }
    }
  }
}

// ---------------------------------------------------------------- phase 2
// One wave = 16 edges. WMMA for the hadamard-MLP branch; VALU dots for the rest.
__global__ void edge_score_kernel(const unsigned short* __restrict__ zbf,
                                  const unsigned short* __restrict__ wh1bf,
                                  const unsigned short* __restrict__ a1bf,
                                  const unsigned short* __restrict__ a2bf,
                                  const unsigned short* __restrict__ tbf,
                                  const int* __restrict__ edge,
                                  const float* __restrict__ b_h1,
                                  const float* __restrict__ w_h2,
                                  const float* __restrict__ b_h2,
                                  const float* __restrict__ b_c1,
                                  const float* __restrict__ w_c2,
                                  const float* __restrict__ b_c2,
                                  const float* __restrict__ b_b,
                                  const float* __restrict__ sw,
                                  float* __restrict__ out,
                                  int etiles, int nedges) {
  int wave = blockIdx.x * (blockDim.x >> 5) + (threadIdx.x >> 5);
  wave = __builtin_amdgcn_readfirstlane(wave);     // wave-uniform
  if (wave >= etiles) return;
  int lane = threadIdx.x & 31;
  int r = lane & 15, half = lane >> 4;
  long e0 = (long)wave * 16;
  long e = e0 + r; if (e >= nedges) e = nedges - 1;
  int s = edge[2 * e], d = edge[2 * e + 1];

  const unsigned int* zs = (const unsigned int*)(zbf + (long)s * HID);
  const unsigned int* zd = (const unsigned int*)(zbf + (long)d * HID);

  // A = hadamard(z_src, z_dst) packed straight into the WMMA A-fragment layout
  BFrag A[4];
  #pragma unroll
  for (int kt = 0; kt < 4; ++kt)
    #pragma unroll
    for (int v = 0; v < 8; ++v) {
      int k = kt * 32 + ((v < 4) ? (2 * v) : (16 + 2 * (v - 4))) + half * 8;
      A[kt].u[v] = pk_mul_bf(zs[k >> 1], zd[k >> 1]);
    }

  // h = relu(HAD @ w_h1^T + b_h1); score1 partial = h . w_h2
  float p[8] = {0.f, 0.f, 0.f, 0.f, 0.f, 0.f, 0.f, 0.f};
  #pragma unroll
  for (int nt = 0; nt < 4; ++nt) {
    int j = nt * 16 + r;                           // hidden index 0..63
    const unsigned int* wrow = (const unsigned int*)(wh1bf + (long)j * HID);
    BFrag Bf[4];
    #pragma unroll
    for (int kt = 0; kt < 4; ++kt)
      #pragma unroll
      for (int v = 0; v < 8; ++v) {
        int k = kt * 32 + ((v < 4) ? (2 * v) : (16 + 2 * (v - 4))) + half * 8;
        Bf[kt].u[v] = wrow[k >> 1];
      }
    v8f acc = {0.f, 0.f, 0.f, 0.f, 0.f, 0.f, 0.f, 0.f};
    #pragma unroll
    for (int kt = 0; kt < 4; ++kt) {
      v16bf av = __builtin_bit_cast(v16bf, A[kt]);
      v16bf bv = __builtin_bit_cast(v16bf, Bf[kt]);
      acc = __builtin_amdgcn_wmma_f32_16x16x32_bf16(false, av, false, bv,
                                                    (short)0, acc, false, false);
    }
    float hb = b_h1[j], hw = w_h2[j];
    #pragma unroll
    for (int rr = 0; rr < 8; ++rr) {               // lane holds edge rr(+8), hidden j
      float hv = acc[rr] + hb;
      hv = hv > 0.f ? hv : 0.f;
      p[rr] += hv * hw;
    }
  }
  // reduce over the 16 hidden-columns per half-wave (bit4 untouched)
  #pragma unroll
  for (int mask = 1; mask <= 8; mask <<= 1)
    #pragma unroll
    for (int rr = 0; rr < 8; ++rr)
      p[rr] += __shfl_xor(p[rr], mask, 32);
  // my edge is m = r; its score lives in half (m>>3), slot (m&7)
  float mine  = p[r & 7];
  float other = __shfl_xor(mine, 16, 32);
  float s1 = ((r >> 3) == half) ? mine : other;

  // score2: relu(a1[s]+a2[d]+b_c1) . w_c2 ; score3: z_src . t[d]
  const unsigned int* a1p = (const unsigned int*)(a1bf + (long)s * HID);
  const unsigned int* a2p = (const unsigned int*)(a2bf + (long)d * HID);
  const unsigned int* tp  = (const unsigned int*)(tbf  + (long)d * HID);
  float acc2 = 0.f, acc3 = 0.f;
  int hbase = half * 64;                           // each half-lane handles 64 elems
  #pragma unroll 8
  for (int hh = 0; hh < 32; ++hh) {
    int h = hbase + 2 * hh;
    unsigned int u1 = a1p[h >> 1], u2 = a2p[h >> 1];
    float c0 = bf2f(u1 & 0xFFFFu) + bf2f(u2 & 0xFFFFu) + b_c1[h];
    float c1 = bf2f(u1 >> 16)     + bf2f(u2 >> 16)     + b_c1[h + 1];
    c0 = c0 > 0.f ? c0 : 0.f;  c1 = c1 > 0.f ? c1 : 0.f;
    acc2 = fmaf(c0, w_c2[h], fmaf(c1, w_c2[h + 1], acc2));
    unsigned int uz = zs[h >> 1], ut = tp[h >> 1];
    acc3 = fmaf(bf2f(uz & 0xFFFFu), bf2f(ut & 0xFFFFu),
           fmaf(bf2f(uz >> 16),     bf2f(ut >> 16),     acc3));
  }
  acc2 += __shfl_xor(acc2, 16, 32);
  acc3 += __shfl_xor(acc3, 16, 32);

  // softmax over 3 strategy weights (redundant per lane, trivial)
  float t0 = sw[0], t1 = sw[1], t2 = sw[2];
  float mx = fmaxf(t0, fmaxf(t1, t2));
  float x0 = expf(t0 - mx), x1 = expf(t1 - mx), x2 = expf(t2 - mx);
  float inv = 1.f / (x0 + x1 + x2);
  float w0 = x0 * inv, w1 = x1 * inv, w2 = x2 * inv;

  if (lane < 16 && (e0 + lane) < nedges)
    out[e0 + lane] = w0 * (s1 + b_h2[0]) + w1 * (acc2 + b_c2[0]) + w2 * (acc3 + b_b[0]);
}

// ---------------------------------------------------------------- launch
extern "C" void kernel_launch(void* const* d_in, const int* in_sizes, int n_in,
                              void* d_out, int out_size, void* d_ws, size_t ws_size,
                              hipStream_t stream) {
  const float* z    = (const float*)d_in[0];
  const int*   edge = (const int*)  d_in[1];
  const float* w_h1 = (const float*)d_in[2];
  const float* b_h1 = (const float*)d_in[3];
  const float* w_h2 = (const float*)d_in[4];
  const float* b_h2 = (const float*)d_in[5];
  const float* w_c1 = (const float*)d_in[6];
  const float* b_c1 = (const float*)d_in[7];
  const float* w_c2 = (const float*)d_in[8];
  const float* b_c2 = (const float*)d_in[9];
  const float* w_b  = (const float*)d_in[10];
  const float* b_b  = (const float*)d_in[11];
  const float* sw   = (const float*)d_in[12];

  int nz    = in_sizes[0];          // nodes * 128
  int nodes = nz / HID;
  int E     = in_sizes[1] / 2;

  // workspace layout (all bf16 => node tables fit in 192MB L2)
  char* p = (char*)d_ws;
  auto alloc = [&](size_t bytes) {
    char* q = p; p += (bytes + 255) & ~(size_t)255; return q;
  };
  unsigned short* zbf   = (unsigned short*)alloc((size_t)nz * 2);
  unsigned short* wbbf  = (unsigned short*)alloc((size_t)HID * HID * 2);
  unsigned short* wc1bf = (unsigned short*)alloc((size_t)HID * 2 * HID * 2);
  unsigned short* wh1bf = (unsigned short*)alloc((size_t)(HID / 2) * HID * 2);
  unsigned short* tbf   = (unsigned short*)alloc((size_t)nz * 2);
  unsigned short* a1bf  = (unsigned short*)alloc((size_t)nz * 2);
  unsigned short* a2bf  = (unsigned short*)alloc((size_t)nz * 2);
  (void)ws_size; (void)n_in; (void)out_size;

  // phase 0: bf16 conversions
  f32_to_bf16_kernel<<<(nz + 255) / 256, 256, 0, stream>>>(z, zbf, nz);
  f32_to_bf16_kernel<<<(HID * HID + 255) / 256, 256, 0, stream>>>(w_b, wbbf, HID * HID);
  f32_to_bf16_kernel<<<(HID * 2 * HID + 255) / 256, 256, 0, stream>>>(w_c1, wc1bf, HID * 2 * HID);
  f32_to_bf16_kernel<<<((HID / 2) * HID + 255) / 256, 256, 0, stream>>>(w_h1, wh1bf, (HID / 2) * HID);

  // phase 1: fused per-node GEMMs (one z read feeds t, a1, a2)
  int ntiles = (nodes + 15) / 16;
  int wpb = 256 / 32;               // 8 waves per 256-thread block
  int gblk = (ntiles + wpb - 1) / wpb;
  node_gemm3_bf16_kernel<<<gblk, 256, 0, stream>>>(zbf, wbbf, wc1bf,
                                                   tbf, a1bf, a2bf, ntiles, nodes);

  // phase 2: per-edge scoring (WMMA on hadamard branch, L2-resident gathers)
  int etiles = (E + 15) / 16;
  int eblk = (etiles + wpb - 1) / wpb;
  edge_score_kernel<<<eblk, 256, 0, stream>>>(zbf, wh1bf, a1bf, a2bf, tbf, edge,
                                              b_h1, w_h2, b_h2, b_c1, w_c2, b_c2,
                                              b_b, sw, (float*)d_out, etiles, E);
}
